// AttentionPool_28123445854479
// MI455X (gfx1250) — matrix operand order repaired and checked by reading, not compiled, not run on hardware
//
#include <hip/hip_runtime.h>
#include <hip/hip_bf16.h>
#include <math.h>

typedef __attribute__((ext_vector_type(16))) __bf16 v16bf;
typedef __attribute__((ext_vector_type(8)))  float  v8f;
typedef __attribute__((ext_vector_type(4)))  float  v4f;

#define C_IN  256
#define C_HID 128
#define KT 8   // 256 / 32  (k-tiles of 32)

// ---------------------------------------------------------------------------
// Kernel 1: scores[i] = tanh(x[i,:] @ W1 + b1) @ W2 + b2 via bf16 WMMA.
// D = A*B, A = W1^T (128x256), B = x^T tile (256x16 nodes).
// Wave w of the block owns hidden tile mt=w and keeps its 8 A-fragments
// (64 VGPRs) resident in registers -> nothing loop-invariant in LDS to hoist,
// no spills. The 16-node x tile is staged to LDS in B-fragment order.
// D layout: lane l<16 -> node l, m=16w+j ; lane l+16 -> node l, m=16w+j+8.
// ---------------------------------------------------------------------------
__global__ void __launch_bounds__(256)
ap_scores_kernel(const float* __restrict__ x,
                 const float* __restrict__ W1,
                 const float* __restrict__ b1,
                 const float* __restrict__ W2,
                 const float* __restrict__ b2,
                 float* __restrict__ scores,
                 int N, int numTiles) {
  __shared__ __bf16 ldsB[KT * 32 * 16];   // 8 KB: x tile in B-fragment order
  __shared__ float  sc[8][16];            // per-wave partial scores

  const int t     = threadIdx.x;
  const int wave  = t >> 5;               // = mt (hidden tile) for this wave
  const int lane  = t & 31;
  const int lhalf = lane & 15;
  const bool hiL  = lane >= 16;
  const int mofs  = hiL ? 8 : 0;

  // --- A fragments (W1^T) in contiguous-k A(16x32) order (k = 16*half + s, as B), registers, once.
  // lane<16: halves 0..15 -> K 0..15 ; lane>=16: K 16..31.
  v16bf aw[KT];
#pragma unroll
  for (int kt = 0; kt < KT; ++kt) {
    const int m = wave * 16 + lhalf;                     // hidden channel (M)
#pragma unroll
    for (int h = 0; h < 16; ++h) {
      const int k = kt * 32 + h + (hiL ? 16 : 0);
      aw[kt][h] = (__bf16)W1[k * C_HID + m];             // W1^T[m,k] = W1[k,m]
    }
  }
  // per-lane b1 / W2 for this lane's output rows
  float b1r[8], w2r[8];
#pragma unroll
  for (int j = 0; j < 8; ++j) {
    const int m = wave * 16 + j + mofs;
    b1r[j] = b1[m];
    w2r[j] = W2[m];
  }
  const float b2v = b2[0];

  // --- staging map: thread t loads 16 contiguous channels of one node and
  // writes one contiguous v16bf fragment chunk in LDS.
  const int sNode = t >> 4;               // 0..15
  const int sC0   = (t & 15) << 4;        // 0,16,...,240
  const int sKt   = sC0 >> 5;
  const int sLane = sNode + (((sC0 & 31) >= 16) ? 16 : 0);
  __bf16* const dst = &ldsB[(sKt * 32 + sLane) << 4];

  const v8f vzero = {0.f, 0.f, 0.f, 0.f, 0.f, 0.f, 0.f, 0.f};

  for (int tile = blockIdx.x; tile < numTiles; tile += gridDim.x) {
    const int rowBase = tile << 4;

    // 1) global load: 16 contiguous floats of one x row (coalesced groups)
    int node = rowBase + sNode;
    if (node >= N) node = N - 1;          // clamp; EXEC must stay full
    const v4f* p4 = (const v4f*)(x + (size_t)node * C_IN + sC0);
    v4f r0 = p4[0], r1 = p4[1], r2 = p4[2], r3 = p4[3];

    __syncthreads();                      // prev iter done reading ldsB / sc

    v16bf bv;
#pragma unroll
    for (int j = 0; j < 4; ++j) {
      bv[j]      = (__bf16)r0[j];
      bv[4 + j]  = (__bf16)r1[j];
      bv[8 + j]  = (__bf16)r2[j];
      bv[12 + j] = (__bf16)r3[j];
    }
    *(v16bf*)dst = bv;
    __syncthreads();

    // 2) 8 WMMAs: D(16x16) += A[mt=wave,kt] * B[kt]
    v8f acc = vzero;
#pragma unroll
    for (int kt = 0; kt < KT; ++kt) {
      v16bf bf = *(const v16bf*)&ldsB[(kt * 32 + lane) << 4];
      acc = __builtin_amdgcn_wmma_f32_16x16x32_bf16(
          false, aw[kt], false, bf, (short)0, acc, false, false);
    }

    // 3) tanh + dot with W2 (branch-free tanh: sign(h)*(1-e)/(1+e), e=exp(-2|h|))
    float partial = 0.f;
#pragma unroll
    for (int j = 0; j < 8; ++j) {
      const float h = acc[j] + b1r[j];
      const float e = __expf(-2.f * fabsf(h));
      const float th = __builtin_copysignf((1.f - e) / (1.f + e), h);
      partial += th * w2r[j];
    }
    partial += __shfl_xor(partial, 16, 32);   // m 0..7 with m 8..15
    if (!hiL) sc[wave][lhalf] = partial;
    __syncthreads();

    // 4) combine the 8 hidden chunks, write scores
    if (t < 16) {
      const int row = rowBase + t;
      if (row < N) {
        float s = b2v;
#pragma unroll
        for (int w = 0; w < 8; ++w) s += sc[w][t];
        scores[row] = s;
      }
    }
  }
}

// ---------------------------------------------------------------------------
// Kernel 2: segment boundaries via binary search (batch is sorted).
// ---------------------------------------------------------------------------
__global__ void ap_bounds_kernel(const int* __restrict__ batch,
                                 int* __restrict__ segStart,
                                 int N, int numGraphs) {
  int g = blockIdx.x * blockDim.x + threadIdx.x;
  if (g > numGraphs) return;
  if (g == numGraphs) { segStart[g] = N; return; }
  int lo = 0, hiq = N;
  while (lo < hiq) {
    int mid = (lo + hiq) >> 1;
    if (batch[mid] < g) lo = mid + 1; else hiq = mid;
  }
  segStart[g] = lo;
}

// ---------------------------------------------------------------------------
// Kernel 3: per-graph softmax -> weights (one 256-thread block per graph).
// ---------------------------------------------------------------------------
__global__ void ap_softmax_kernel(const float* __restrict__ scores,
                                  const int* __restrict__ segStart,
                                  float* __restrict__ weights) {
  __shared__ float wred[8];
  __shared__ float sbcast;
  const int g = blockIdx.x;
  const int s = segStart[g], e = segStart[g + 1];
  const int t = threadIdx.x, lane = t & 31, wave = t >> 5;

  float m = -3.402823466e+38f;
  for (int i = s + t; i < e; i += 256) m = fmaxf(m, scores[i]);
  for (int o = 16; o; o >>= 1) m = fmaxf(m, __shfl_xor(m, o, 32));
  if (lane == 0) wred[wave] = m;
  __syncthreads();
  if (t == 0) {
    float v = wred[0];
    for (int i = 1; i < 8; ++i) v = fmaxf(v, wred[i]);
    sbcast = v;
  }
  __syncthreads();
  const float gmax = sbcast;

  float sum = 0.f;
  for (int i = s + t; i < e; i += 256) sum += expf(scores[i] - gmax);
  for (int o = 16; o; o >>= 1) sum += __shfl_xor(sum, o, 32);
  __syncthreads();                 // wred reuse: everyone is past the read
  if (lane == 0) wred[wave] = sum;
  __syncthreads();
  if (t == 0) {
    float v = 0.f;
    for (int i = 0; i < 8; ++i) v += wred[i];
    sbcast = v + 1e-16f;
  }
  __syncthreads();
  const float inv = 1.0f / sbcast;

  for (int i = s + t; i < e; i += 256) weights[i] = expf(scores[i] - gmax) * inv;
}

// ---------------------------------------------------------------------------
// Kernel 4: weighted segment sum. Block per graph, thread = channel (fully
// coalesced 1KB row reads); x streamed once.
// ---------------------------------------------------------------------------
__global__ void ap_pool_kernel(const float* __restrict__ x,
                               const float* __restrict__ weights,
                               const int* __restrict__ segStart,
                               float* __restrict__ out) {
  const int g = blockIdx.x;
  const int c = threadIdx.x;          // 0..255
  const int s = segStart[g], e = segStart[g + 1];
  float acc = 0.f;
  for (int i = s; i < e; ++i)
    acc = fmaf(x[(size_t)i * C_IN + c], weights[i], acc);
  out[(size_t)g * C_IN + c] = acc;    // empty segment -> 0, matches reference
}

// ---------------------------------------------------------------------------
extern "C" void kernel_launch(void* const* d_in, const int* in_sizes, int n_in,
                              void* d_out, int out_size, void* d_ws, size_t ws_size,
                              hipStream_t stream) {
  const float* x     = (const float*)d_in[0];
  const int*   batch = (const int*)  d_in[1];
  const float* W1    = (const float*)d_in[2];
  const float* b1    = (const float*)d_in[3];
  const float* W2    = (const float*)d_in[4];
  const float* b2    = (const float*)d_in[5];
  float* out = (float*)d_out;

  const int N         = in_sizes[1];        // N_NODES
  const int numGraphs = out_size / C_IN;    // 4096

  float* scores   = (float*)d_ws;           // N floats
  float* weights  = scores + N;             // N floats
  int*   segStart = (int*)(weights + N);    // numGraphs + 1 ints

  const int numTiles = (N + 15) / 16;
  int grid1 = numTiles < 2048 ? numTiles : 2048;

  ap_scores_kernel<<<grid1, 256, 0, stream>>>(x, W1, b1, W2, b2, scores, N, numTiles);
  ap_bounds_kernel<<<(numGraphs + 256) / 256, 256, 0, stream>>>(batch, segStart, N, numGraphs);
  ap_softmax_kernel<<<numGraphs, 256, 0, stream>>>(scores, segStart, weights);
  ap_pool_kernel<<<numGraphs, 256, 0, stream>>>(x, weights, segStart, out);
}
